// MPlusRetrieverLightweight_37830071943375
// MI455X (gfx1250) — matrix operand order repaired and checked by reading, not compiled, not run on hardware
//
#include <hip/hip_runtime.h>
#include <hip/hip_bf16.h>

// ---------------------------------------------------------------------------
// MPlusRetrieverLightweight for MI455X (gfx1250, wave32, WMMA)
//
// Dataflow: gather+pool+LN (HBM bound, ~0.5-1.07 GB) -> f16 WMMA GEMM1
// (272x4096 @ 4096x1024) -> f16 WMMA GEMM2 (272x1024 @ 1024x1024) -> scores
// -> top-k. The query row is padded into M-tile 16 (rows 257..271 zero) so it
// shares the WMMA GEMM path with per-tile q-weight selection.
// ---------------------------------------------------------------------------

typedef __attribute__((ext_vector_type(16))) _Float16 v16h;
typedef __attribute__((ext_vector_type(8)))  _Float16 v8h;
typedef __attribute__((ext_vector_type(8)))  float    v8f;
typedef __attribute__((ext_vector_type(4)))  float    v4f;

#define D_EMB 4096
#define HD    1024
#define SEQ   256
#define NKEY  256
#define MROWS 272   // 256 key rows + 16 padded query rows (row 256 = query)

// ---------------------------------------------------------------------------
// Kernel 1: embedding gather + masked mean pool + LayerNorm -> f32 X[MROWS,D]
// One block per output row; 256 threads, 16 contiguous f32 columns per thread.
// ---------------------------------------------------------------------------
__global__ __launch_bounds__(256)
void pool_ln_kernel(const int* __restrict__ input_ids,
                    const int* __restrict__ attn_mask,
                    const int* __restrict__ query_ids,
                    const int* __restrict__ query_mask,
                    const float* __restrict__ embed,
                    const float* __restrict__ ln_g,
                    const float* __restrict__ ln_b,
                    float* __restrict__ X)
{
    const int b = blockIdx.x;
    const int t = threadIdx.x;
    float* xrow = X + (size_t)b * D_EMB;

    if (b > NKEY) {                    // zero padding rows 257..271
        v4f z = {0.f, 0.f, 0.f, 0.f};
        v4f* p = (v4f*)(xrow + t * 16);
        p[0] = z; p[1] = z; p[2] = z; p[3] = z;
        return;
    }

    __shared__ int   s_ids[SEQ];
    __shared__ int   s_cnt;
    __shared__ float red[256];

    const int* ids = (b < NKEY) ? (input_ids + b * SEQ) : query_ids;
    const int* msk = (b < NKEY) ? (attn_mask + b * SEQ) : query_mask;

    if (t == 0) {                      // compact masked tokens: halves gather BW
        int c = 0;
        for (int s = 0; s < SEQ; ++s)
            if (msk[s] != 0) s_ids[c++] = ids[s];
        s_cnt = c;
    }
    __syncthreads();
    const int cnt = s_cnt;

    float acc[16];
    #pragma unroll
    for (int i = 0; i < 16; ++i) acc[i] = 0.f;

    const int col = t * 16;
    for (int s = 0; s < cnt; ++s) {
        const v4f* r = (const v4f*)(embed + (size_t)s_ids[s] * D_EMB + col);
        v4f r0 = r[0], r1 = r[1], r2 = r[2], r3 = r[3];
        #pragma unroll
        for (int i = 0; i < 4; ++i) {
            acc[i]      += r0[i];
            acc[4 + i]  += r1[i];
            acc[8 + i]  += r2[i];
            acc[12 + i] += r3[i];
        }
    }
    const float inv = 1.0f / ((float)cnt + 1e-9f);
    float ls = 0.f, lq = 0.f;
    #pragma unroll
    for (int i = 0; i < 16; ++i) {
        acc[i] *= inv;
        ls += acc[i];
        lq += acc[i] * acc[i];
    }

    // block reduction: mean
    red[t] = ls; __syncthreads();
    #pragma unroll
    for (int off = 128; off > 0; off >>= 1) {
        if (t < off) red[t] += red[t + off];
        __syncthreads();
    }
    const float mu = red[0] * (1.0f / (float)D_EMB);
    __syncthreads();
    // block reduction: E[x^2]
    red[t] = lq; __syncthreads();
    #pragma unroll
    for (int off = 128; off > 0; off >>= 1) {
        if (t < off) red[t] += red[t + off];
        __syncthreads();
    }
    const float var  = red[0] * (1.0f / (float)D_EMB) - mu * mu;
    const float rstd = rsqrtf(var + 1e-5f);

    #pragma unroll
    for (int i = 0; i < 16; ++i) {
        const int c2 = col + i;
        xrow[c2] = (acc[i] - mu) * rstd * ln_g[c2] + ln_b[c2];
    }
}

// ---------------------------------------------------------------------------
// f32 -> f16 convert-on-load helpers matching WMMA lane layouts.
// B tile (KxN): per lane, K chunks [0..7] and [16..23] from p (W stored [N_out, K] row-major).
// ---------------------------------------------------------------------------
__device__ inline v16h cvt16f(const float* __restrict__ p)
{
    const v4f* q = (const v4f*)p;
    v4f x0 = q[0], x1 = q[1], x2 = q[4], x3 = q[5];
    v16h h;
    #pragma unroll
    for (int i = 0; i < 4; ++i) {
        h[i]      = (_Float16)x0[i];
        h[4 + i]  = (_Float16)x1[i];
        h[8 + i]  = (_Float16)x2[i];
        h[12 + i] = (_Float16)x3[i];
    }
    return h;
}

// ---------------------------------------------------------------------------
// Kernel 2: H1 = relu(X @ W1^T + b1) in f16.  One wave per block; each wave
// computes one 16-row M-tile x four 16-col N-tiles.  K = 4096 (128 WMMA steps).
// M-tile 16 uses the query weights; the rest use the key weights.
// ---------------------------------------------------------------------------
__global__ __launch_bounds__(32)
void gemm1_kernel(const float* __restrict__ X,
                  const float* __restrict__ kW1, const float* __restrict__ kb1,
                  const float* __restrict__ qW1, const float* __restrict__ qb1,
                  _Float16* __restrict__ H1)
{
    const int lane  = threadIdx.x;
    const int hg    = lane >> 4;      // half-wave group (0 or 1)
    const int lid   = lane & 15;
    const int mtile = blockIdx.y;     // 0..16
    const int ng0   = blockIdx.x * 4; // 4 N-tiles per wave -> reuse A regs

    const float* W    = (mtile == 16) ? qW1 : kW1;
    const float* bias = (mtile == 16) ? qb1 : kb1;

    const int    m    = mtile * 16 + lid;          // A: lanes 0-15 / 16-31 both hold M=0..15
    const float* arow = X + (size_t)m * D_EMB;

    const float* brow[4];
    #pragma unroll
    for (int j = 0; j < 4; ++j) {
        const int n = (ng0 + j) * 16 + lid;        // B: N = lane&15
        brow[j] = W + (size_t)n * D_EMB + 8 * hg;  // K chunks [8*hg..+7], [16+8*hg..+7]
    }

    v8f c[4] = {};

    for (int k0 = 0; k0 < D_EMB; k0 += 32) {
        // A 16x32 f16 layout: lane holds K chunks [8*hg .. +7] and [16+8*hg .. +7]
        v16h a;
        {
            const v4f* p1 = (const v4f*)(arow + k0 + 8 * hg);
            const v4f* p2 = (const v4f*)(arow + k0 + 16 + 8 * hg);
            v4f x0 = p1[0], x1 = p1[1], x2 = p2[0], x3 = p2[1];
            #pragma unroll
            for (int i = 0; i < 4; ++i) {
                a[i]      = (_Float16)x0[i];
                a[4 + i]  = (_Float16)x1[i];
                a[8 + i]  = (_Float16)x2[i];
                a[12 + i] = (_Float16)x3[i];
            }
        }
        #pragma unroll
        for (int j = 0; j < 4; ++j) {
            v16h bm = cvt16f(brow[j] + k0);
            c[j] = __builtin_amdgcn_wmma_f32_16x16x32_f16(
                false, a, false, bm, (short)0, c[j], false, false);
        }
    }

    // epilogue: bias + ReLU, store f16.  C/D: row r -> M = r + 8*hg, N = lane&15
    #pragma unroll
    for (int j = 0; j < 4; ++j) {
        const int   n  = (ng0 + j) * 16 + lid;
        const float bv = bias[n];
        #pragma unroll
        for (int r = 0; r < 8; ++r) {
            const int mm = mtile * 16 + r + 8 * hg;
            float v = c[j][r] + bv;
            v = v > 0.f ? v : 0.f;
            H1[(size_t)mm * HD + n] = (_Float16)v;
        }
    }
}

// ---------------------------------------------------------------------------
// Kernel 3: H2 = H1 @ W2^T + b2 in f32.  Same tiling; A already f16.
// ---------------------------------------------------------------------------
__global__ __launch_bounds__(32)
void gemm2_kernel(const _Float16* __restrict__ H1,
                  const float* __restrict__ kW2, const float* __restrict__ kb2,
                  const float* __restrict__ qW2, const float* __restrict__ qb2,
                  float* __restrict__ H2)
{
    const int lane  = threadIdx.x;
    const int hg    = lane >> 4;
    const int lid   = lane & 15;
    const int mtile = blockIdx.y;
    const int ng0   = blockIdx.x * 4;

    const float* W    = (mtile == 16) ? qW2 : kW2;
    const float* bias = (mtile == 16) ? qb2 : kb2;

    const int        m    = mtile * 16 + lid;
    const _Float16*  arow = H1 + (size_t)m * HD;

    const float* brow[4];
    #pragma unroll
    for (int j = 0; j < 4; ++j) {
        const int n = (ng0 + j) * 16 + lid;
        brow[j] = W + (size_t)n * HD + 8 * hg;
    }

    v8f c[4] = {};

    for (int k0 = 0; k0 < HD; k0 += 32) {
        v16h a;
        {
            v8h lo = *(const v8h*)(arow + k0 + 8 * hg);
            v8h hi = *(const v8h*)(arow + k0 + 16 + 8 * hg);
            #pragma unroll
            for (int i = 0; i < 8; ++i) { a[i] = lo[i]; a[8 + i] = hi[i]; }
        }
        #pragma unroll
        for (int j = 0; j < 4; ++j) {
            v16h bm = cvt16f(brow[j] + k0);
            c[j] = __builtin_amdgcn_wmma_f32_16x16x32_f16(
                false, a, false, bm, (short)0, c[j], false, false);
        }
    }

    #pragma unroll
    for (int j = 0; j < 4; ++j) {
        const int   n  = (ng0 + j) * 16 + lid;
        const float bv = bias[n];
        #pragma unroll
        for (int r = 0; r < 8; ++r) {
            const int mm = mtile * 16 + r + 8 * hg;
            H2[(size_t)mm * HD + n] = c[j][r] + bv;
        }
    }
}

// ---------------------------------------------------------------------------
// Kernel 4: scores[b] = sigmoid(dot(query, keys[b])).  One wave per row.
// ---------------------------------------------------------------------------
__global__ __launch_bounds__(256)
void score_kernel(const float* __restrict__ H2, float* __restrict__ sc)
{
    const int wave = threadIdx.x >> 5;
    const int lane = threadIdx.x & 31;
    const int b    = blockIdx.x * 8 + wave;

    const float* q  = H2 + (size_t)NKEY * HD;   // row 256 = query
    const float* kr = H2 + (size_t)b * HD;

    float s = 0.f;
    for (int h = lane; h < HD; h += 32) s += q[h] * kr[h];
    #pragma unroll
    for (int off = 16; off > 0; off >>= 1) s += __shfl_down(s, off, 32);
    if (lane == 0) sc[b] = 1.0f / (1.0f + __expf(-s));
}

// ---------------------------------------------------------------------------
// Kernel 5: top-k via iterative block argmax (k is tiny).  Output layout:
// out[0..k-1] = values, out[k..2k-1] = indices (as float, exact for idx<2^24).
// ---------------------------------------------------------------------------
__global__ __launch_bounds__(256)
void topk_kernel(const float* __restrict__ sc, const int* __restrict__ kptr,
                 float* __restrict__ out)
{
    __shared__ float sv[256];
    __shared__ float wv[256];
    __shared__ int   wi[256];
    const int t = threadIdx.x;
    sv[t] = sc[t];
    __syncthreads();
    const int k = kptr[0];
    for (int it = 0; it < k; ++it) {
        wv[t] = sv[t]; wi[t] = t;
        __syncthreads();
        #pragma unroll
        for (int off = 128; off > 0; off >>= 1) {
            if (t < off) {
                if (wv[t + off] > wv[t]) { wv[t] = wv[t + off]; wi[t] = wi[t + off]; }
            }
            __syncthreads();
        }
        if (t == 0) {
            out[it]     = wv[0];
            out[k + it] = (float)wi[0];
            sv[wi[0]]   = -3.402823466e38f;
        }
        __syncthreads();
    }
}

// ---------------------------------------------------------------------------
extern "C" void kernel_launch(void* const* d_in, const int* in_sizes, int n_in,
                              void* d_out, int out_size, void* d_ws, size_t ws_size,
                              hipStream_t stream)
{
    const int*   input_ids  = (const int*)d_in[0];
    const int*   attn_mask  = (const int*)d_in[1];
    const int*   query_ids  = (const int*)d_in[2];
    const int*   query_mask = (const int*)d_in[3];
    const float* embed      = (const float*)d_in[4];
    const float* ln_g       = (const float*)d_in[5];
    const float* ln_b       = (const float*)d_in[6];
    const float* kW1        = (const float*)d_in[7];
    const float* kb1        = (const float*)d_in[8];
    const float* kW2        = (const float*)d_in[9];
    const float* kb2        = (const float*)d_in[10];
    const float* qW1        = (const float*)d_in[11];
    const float* qb1        = (const float*)d_in[12];
    const float* qW2        = (const float*)d_in[13];
    const float* qb2        = (const float*)d_in[14];
    const int*   kptr       = (const int*)d_in[15];

    // workspace layout (~6.13 MB total)
    char* ws = (char*)d_ws;
    const size_t X_OFF  = 0;
    const size_t H1_OFF = X_OFF  + (size_t)MROWS * D_EMB * sizeof(float);    // 4,456,448
    const size_t H2_OFF = H1_OFF + (size_t)MROWS * HD   * sizeof(_Float16);  // +557,056
    const size_t SC_OFF = H2_OFF + (size_t)MROWS * HD   * sizeof(float);     // +1,114,112
    float*     X  = (float*)(ws + X_OFF);
    _Float16*  H1 = (_Float16*)(ws + H1_OFF);
    float*     H2 = (float*)(ws + H2_OFF);
    float*     SC = (float*)(ws + SC_OFF);

    pool_ln_kernel<<<MROWS, 256, 0, stream>>>(input_ids, attn_mask, query_ids,
                                              query_mask, embed, ln_g, ln_b, X);
    gemm1_kernel<<<dim3(HD / 64, MROWS / 16), 32, 0, stream>>>(X, kW1, kb1,
                                                               qW1, qb1, H1);
    gemm2_kernel<<<dim3(HD / 64, MROWS / 16), 32, 0, stream>>>(H1, kW2, kb2,
                                                               qW2, qb2, H2);
    score_kernel<<<NKEY / 8, 256, 0, stream>>>(H2, SC);
    topk_kernel<<<1, 256, 0, stream>>>(SC, kptr, (float*)d_out);
}
